// Transformer_Layer_48473000903646
// MI455X (gfx1250) — matrix operand order repaired and checked by reading, not compiled, not run on hardware
//
#include <hip/hip_runtime.h>
#include <hip/hip_bf16.h>
#include <math.h>

// ---------------- problem constants ----------------
#define B_   32
#define PN_  16
#define PS_  8
#define D_   128
#define N_   32
#define DFF_ 512
#define DP_  1024   // D*PS
#define SEQ_ 128    // PN*PS

typedef __attribute__((ext_vector_type(16))) _Float16 v16h_t;
typedef __attribute__((ext_vector_type(8)))  _Float16 v8h_t;
typedef __attribute__((ext_vector_type(8)))  float    v8f_t;

// ---------------- CDNA5 async global->LDS (guarded) ----------------
#if defined(__HIP_DEVICE_COMPILE__) && defined(__gfx1250__) && \
    __has_builtin(__builtin_amdgcn_global_load_async_to_lds_b128)
#define USE_ASYNC_LDS 1
// Builtin signature (probe-verified): arg0 = int4 addrspace(1)*,
// arg1 = int4 addrspace(3)*, then imm offset, imm cpol.
typedef __attribute__((__vector_size__(16))) int v4i_t;
typedef __attribute__((address_space(1))) v4i_t* g4p_t;
typedef __attribute__((address_space(3))) v4i_t* l4p_t;
#else
#define USE_ASYNC_LDS 0
#endif

template <int NREM>
__device__ __forceinline__ void wait_async_n() {
#if USE_ASYNC_LDS
#if __has_builtin(__builtin_amdgcn_s_wait_asynccnt)
  __builtin_amdgcn_s_wait_asynccnt(NREM);
#else
  asm volatile("s_wait_asynccnt %0" ::"i"(NREM) : "memory");
#endif
#endif
}

// copy 16 halves (32B) global -> LDS (async DMA when available)
__device__ __forceinline__ void ld_tile16(_Float16* dst, const _Float16* src) {
#if USE_ASYNC_LDS
  __builtin_amdgcn_global_load_async_to_lds_b128(
      (g4p_t)(void*)const_cast<_Float16*>(src), (l4p_t)(void*)dst, 0, 0);
  __builtin_amdgcn_global_load_async_to_lds_b128(
      (g4p_t)(void*)const_cast<_Float16*>(src + 8), (l4p_t)(void*)(dst + 8), 0, 0);
#else
  *(v8h_t*)dst       = *(const v8h_t*)src;
  *(v8h_t*)(dst + 8) = *(const v8h_t*)(src + 8);
#endif
}

// ==================================================================
// Generic tiled WMMA GEMM:  C[z] = act(A[z] @ B[z] + bias[z] + row_add)
//   f16 inputs, f32 accumulate via v_wmma_f32_16x16x32_f16.
//   Block = 256 threads = 8 wave32; tile 128x128, K-step 32.
//   Wave grid 4x2 -> each wave owns 32x64 = 2x4 fragments of 16x16.
//   Double-buffered LDS: async DMA of tile k+1 overlaps WMMA of tile k
//   (each thread has 4 in-flight async ops per tile; loads complete in
//   order, so s_wait_asynccnt 4 retires exactly the current tile).
//   ACT: 0=none 1=relu 2=gelu(exact).  OHALF: 1 -> f16 out, 0 -> f32 out.
// ==================================================================
template <int ACT, int OHALF>
__global__ __launch_bounds__(256) void k_wmma_gemm(
    const _Float16* __restrict__ A, long a_bs, int lda,
    const _Float16* __restrict__ Bm, long b_bs, int ldb,
    const float* __restrict__ bias, long bias_bs,
    const float* __restrict__ row_add, int row_mod,
    void* __restrict__ Cout, long c_bs, int ldc,
    int M, int Ncols, int K) {
  // LDS: A tile 128x32 (row stride 40 halves = 80B, 16B aligned),
  //      B tile 32x128 (row stride 144 halves = 288B, 32B aligned), x2 buffers
  __shared__ _Float16 sA[2][128][40];
  __shared__ _Float16 sB[2][32][144];

  const int tid  = threadIdx.x;
  const int lane = tid & 31;
  const int wave = tid >> 5;   // 0..7
  const int wm   = wave >> 1;  // 0..3 : 32-row slice
  const int wn   = wave & 1;   // 0..1 : 64-col slice
  const int tileM = blockIdx.y * 128;
  const int tileN = blockIdx.x * 128;
  const int z     = blockIdx.z;

  const _Float16* Ab = A  + (long)z * a_bs;
  const _Float16* Bb = Bm + (long)z * b_bs;

  const v8f_t zero = {0.f, 0.f, 0.f, 0.f, 0.f, 0.f, 0.f, 0.f};
  v8f_t acc[2][4];
#pragma unroll
  for (int i = 0; i < 2; ++i)
#pragma unroll
    for (int j = 0; j < 4; ++j) acc[i][j] = zero;

  // cooperative tile-fill assignments (16 halves per thread per tile)
  const int ar = tid >> 1, ac = (tid & 1) * 16;  // A: 128 rows x 32 cols
  const int br = tid >> 3, bc = (tid & 7) * 16;  // B:  32 rows x 128 cols

  const int g    = lane >> 4;  // A-fragment lane group (K split per ISA layout)
  const int rrow = lane & 15;  // A-fragment row

  auto issue_tile = [&](int k0, int buf) {
    const _Float16* ap = Ab + (long)(tileM + ar) * lda + (k0 + ac);
    const _Float16* bp = Bb + (long)(k0 + br) * ldb + (tileN + bc);
    ld_tile16(&sA[buf][ar][ac], ap);
    ld_tile16(&sB[buf][br][bc], bp);
  };

  issue_tile(0, 0);  // prologue

  for (int k0 = 0; k0 < K; k0 += 32) {
    const int cur = (k0 >> 5) & 1;
    if (k0 + 32 < K) {
      issue_tile(k0 + 32, cur ^ 1);  // overlap DMA of next tile with compute
      if (k0 + 64 < K) {             // keep L2 warm 2 tiles ahead
        __builtin_prefetch(Ab + (long)(tileM + ar) * lda + (k0 + 64 + ac), 0, 1);
        __builtin_prefetch(Bb + (long)(k0 + 64 + br) * ldb + (tileN + bc), 0, 1);
      }
      wait_async_n<4>();  // retire current tile, next tile still in flight
    } else {
      wait_async_n<0>();
    }
    __syncthreads();

    // A fragments: lane group 0 holds K {0..7,16..23}, group 1 {8..15,24..31}
    v16h_t afrag[2];
#pragma unroll
    for (int i = 0; i < 2; ++i) {
      const _Float16* as = &sA[cur][wm * 32 + i * 16 + rrow][0];
      v8h_t lo = *(const v8h_t*)(as + g * 8);
      v8h_t hi = *(const v8h_t*)(as + 16 + g * 8);
      afrag[i] = __builtin_shufflevector(lo, hi, 0, 1, 2, 3, 4, 5, 6, 7, 8, 9,
                                         10, 11, 12, 13, 14, 15);
    }
#pragma unroll
    for (int j = 0; j < 4; ++j) {
      // B fragment: column N = lane&15, K in same order as A fragment
      v16h_t bfrag;
#pragma unroll
      for (int s = 0; s < 16; ++s) {
        const int kk = (s < 8) ? (g * 8 + s) : (16 + g * 8 + (s - 8));
        bfrag[s] = sB[cur][kk][wn * 64 + j * 16 + rrow];
      }
#pragma unroll
      for (int i = 0; i < 2; ++i) {
        acc[i][j] = __builtin_amdgcn_wmma_f32_16x16x32_f16(
            false, afrag[i], false, bfrag, (short)0, acc[i][j], false, false);
      }
    }
    __syncthreads();  // protect buffer reuse at iteration k0+64
  }

  // Epilogue. C/D layout: VGPR r, lane l -> M = r + 8*(l>>4), N = l&15.
  const int nl = lane & 15;
  const int mh = lane >> 4;
  const float* biasb = bias ? bias + (long)z * bias_bs : nullptr;
#pragma unroll
  for (int i = 0; i < 2; ++i) {
#pragma unroll
    for (int j = 0; j < 4; ++j) {
#pragma unroll
      for (int r = 0; r < 8; ++r) {
        const int m = tileM + wm * 32 + i * 16 + r + 8 * mh;
        const int n = tileN + wn * 64 + j * 16 + nl;
        float v = acc[i][j][r];
        if (biasb) v += biasb[n];
        if (row_add) v += row_add[(m % row_mod) * Ncols + n];
        if (ACT == 1) v = fmaxf(v, 0.0f);
        if (ACT == 2) v = 0.5f * v * (1.0f + erff(v * 0.70710678118654752f));
        const long o = (long)z * c_bs + (long)m * ldc + n;
        if (OHALF) ((_Float16*)Cout)[o] = (_Float16)v;
        else       ((float*)Cout)[o]    = v;
      }
    }
  }
  (void)M;
}

// ---------------- f32 -> f16 convert ----------------
__global__ void k_cvt_f16(const float* __restrict__ s, _Float16* __restrict__ d,
                          long n) {
  long i  = (long)blockIdx.x * blockDim.x + threadIdx.x;
  long st = (long)gridDim.x * blockDim.x;
  for (; i < n; i += st) d[i] = (_Float16)s[i];
}

// ---------------- memory MLP + bias_d + emb ----------------
__global__ __launch_bounds__(256) void k_prep_mem(
    const float* __restrict__ memory, const float* __restrict__ g1w,
    const float* __restrict__ g1b, const float* __restrict__ g2w,
    const float* __restrict__ g2b, const float* __restrict__ g3w,
    const float* __restrict__ g3b, const float* __restrict__ Bd,
    const float* __restrict__ ie, const float* __restrict__ egw,
    const float* __restrict__ egb, float* __restrict__ mem_out,
    float* __restrict__ bias_d, _Float16* __restrict__ emb_h) {
  __shared__ float h1[32][64];
  __shared__ float h2[32][64];
  __shared__ float mm[32][100];
  const int tid = threadIdx.x;
  for (int idx = tid; idx < 32 * 64; idx += 256) {
    int nn = idx >> 6, c = idx & 63;
    float s = g1b[c];
    for (int m = 0; m < 16; ++m) s += memory[nn * 16 + m] * g1w[m * 64 + c];
    h1[nn][c] = tanhf(s);
  }
  __syncthreads();
  for (int idx = tid; idx < 32 * 64; idx += 256) {
    int nn = idx >> 6, c = idx & 63;
    float s = g2b[c];
    for (int m = 0; m < 64; ++m) s += h1[nn][m] * g2w[m * 64 + c];
    h2[nn][c] = tanhf(s);
  }
  __syncthreads();
  for (int idx = tid; idx < 32 * 100; idx += 256) {
    int nn = idx / 100, c = idx % 100;
    float s = g3b[c];
    for (int m = 0; m < 64; ++m) s += h2[nn][m] * g3w[m * 100 + c];
    mm[nn][c]    = s;
    mem_out[idx] = s;
  }
  __syncthreads();
  // bias_d[w][n][d] = sum_m mem[n][m] * Bd[w][m][d]
  for (int idx = tid; idx < 2 * 32 * 128; idx += 256) {
    int w = idx >> 12, nn = (idx >> 7) & 31, dd = idx & 127;
    float s = 0.f;
    for (int m = 0; m < 100; ++m) s += mm[nn][m] * Bd[(w * 100 + m) * 128 + dd];
    bias_d[idx] = s;
  }
  // emb[p][n][d] = sum_m ie[p][n][m] * egw[p][m][d] + egb[p][d]
  for (int idx = tid; idx < 16 * 32 * 128; idx += 256) {
    int p = idx >> 12, nn = (idx >> 7) & 31, dd = idx & 127;
    float s = egb[p * 128 + dd];
    for (int m = 0; m < 16; ++m)
      s += ie[(p * 32 + nn) * 16 + m] * egw[(p * 16 + m) * 128 + dd];
    emb_h[idx] = (_Float16)s;
  }
}

// ---------------- Wd[w][n] = Pd[w] @ memM[n] @ Qd[w]  (f16 out) ----------------
__global__ __launch_bounds__(256) void k_prep_wd(const float* __restrict__ Pd,
                                                 const float* __restrict__ Qd,
                                                 const float* __restrict__ mem,
                                                 _Float16* __restrict__ Wd_h) {
  const int w = blockIdx.x >> 5, nn = blockIdx.x & 31;
  __shared__ float pm[128][10];
  __shared__ float mloc[100];
  const int tid = threadIdx.x;
  for (int i = tid; i < 100; i += 256) mloc[i] = mem[nn * 100 + i];
  __syncthreads();
  for (int idx = tid; idx < 1280; idx += 256) {
    int dd = idx / 10, kk = idx % 10;
    float s = 0.f;
    for (int m = 0; m < 10; ++m) s += Pd[(w * 128 + dd) * 10 + m] * mloc[m * 10 + kk];
    pm[dd][kk] = s;
  }
  __syncthreads();
  for (int idx = tid; idx < 128 * 128; idx += 256) {
    int dd = idx >> 7, e = idx & 127;
    float s = 0.f;
    for (int k = 0; k < 10; ++k) s += pm[dd][k] * Qd[(w * 10 + k) * 128 + e];
    Wd_h[(long)(w * 32 + nn) * 16384 + idx] = (_Float16)s;
  }
}

// ---------------- build t (p,b,tt,n,d) f16 : tt=0 -> emb, else x slice ----------------
__global__ void k_build_t(const float* __restrict__ x,
                          const _Float16* __restrict__ emb_h,
                          _Float16* __restrict__ t_h, long total) {
  long i  = (long)blockIdx.x * blockDim.x + threadIdx.x;
  long st = (long)gridDim.x * blockDim.x;
  for (; i < total; i += st) {
    int dd = (int)(i & 127);
    long r = i >> 7;
    int nn = (int)(r & 31); r >>= 5;
    int tt = (int)(r % 9);  r /= 9;
    int b  = (int)(r & 31);
    int p  = (int)(r >> 5);
    _Float16 v;
    if (tt == 0) v = emb_h[(p * 32 + nn) * 128 + dd];
    else {
      int s = p * 8 + (tt - 1);
      v = (_Float16)x[(((long)b * 128 + s) * 32 + nn) * 128 + dd];
    }
    t_h[i] = v;
  }
}

// ---------------- pack xr (b*N, PN, DP) f16 with DP = d*8 + s ----------------
__global__ void k_pack_xr(const float* __restrict__ x, _Float16* __restrict__ xr,
                          long total) {
  long i  = (long)blockIdx.x * blockDim.x + threadIdx.x;
  long st = (long)gridDim.x * blockDim.x;
  for (; i < total; i += st) {
    int dp = (int)(i & 1023);
    int p  = (int)((i >> 10) & 15);
    long t = i >> 14;
    int nn = (int)(t & 31);
    int b  = (int)(t >> 5);
    int s  = p * 8 + (dp & 7);
    int d  = dp >> 3;
    xr[i] = (_Float16)x[(((long)b * 128 + s) * 32 + nn) * 128 + d];
  }
}

// ---------------- intra attention: 1 query, 9 keys, 2 heads of 64 ----------------
__global__ void k_intra_attn(const _Float16* __restrict__ emb_h,
                             const _Float16* __restrict__ kbuf,
                             const _Float16* __restrict__ vbuf,
                             _Float16* __restrict__ xa_h) {
  int idx = blockIdx.x * blockDim.x + threadIdx.x;  // (p,b,n,h)
  if (idx >= 16 * 32 * 32 * 2) return;
  const int h  = idx & 1;
  const int nn = (idx >> 1) & 31;
  const int b  = (idx >> 6) & 31;
  const int p  = idx >> 11;
  const _Float16* q = &emb_h[(p * 32 + nn) * 128 + h * 64];
  const long base = (long)(p * 32 + b) * 9 * 4096 + nn * 128 + h * 64;
  float sc[9], mx = -1e30f;
  for (int j = 0; j < 9; ++j) {
    const _Float16* kk = kbuf + base + (long)j * 4096;
    float s = 0.f;
    for (int e = 0; e < 64; ++e) s += (float)q[e] * (float)kk[e];
    s *= 0.125f;  // 1/sqrt(64)
    sc[j] = s;
    mx = fmaxf(mx, s);
  }
  float den = 0.f;
  for (int j = 0; j < 9; ++j) { sc[j] = expf(sc[j] - mx); den += sc[j]; }
  const float inv = 1.0f / den;
  _Float16* out = &xa_h[((long)(p * 32 + b) * 32 + nn) * 128 + h * 64];
  for (int e = 0; e < 64; ++e) {
    float o = 0.f;
    for (int j = 0; j < 9; ++j) o += sc[j] * (float)vbuf[base + (long)j * 4096 + e];
    out[e] = (_Float16)(o * inv);
  }
}

// ---------------- inter attention: 128 heads of dim 8 over PN=16 ----------------
__global__ void k_inter_attn(const _Float16* __restrict__ qb,
                             const _Float16* __restrict__ kb,
                             const _Float16* __restrict__ vb,
                             _Float16* __restrict__ ob) {
  long idx = (long)blockIdx.x * blockDim.x + threadIdx.x;  // (s,h,q)
  if (idx >= (long)1024 * 128 * 16) return;
  const int qi = (int)(idx & 15);
  const int h  = (int)((idx >> 4) & 127);
  const int s  = (int)(idx >> 11);
  const long rb = (long)s * 16 * 1024 + h * 8;
  float qv[8];
  for (int e = 0; e < 8; ++e) qv[e] = (float)qb[rb + (long)qi * 1024 + e];
  float sc[16], mx = -1e30f;
  for (int j = 0; j < 16; ++j) {
    float t = 0.f;
    for (int e = 0; e < 8; ++e) t += qv[e] * (float)kb[rb + (long)j * 1024 + e];
    t *= 0.35355339059327373f;  // 1/sqrt(8)
    sc[j] = t;
    mx = fmaxf(mx, t);
  }
  float den = 0.f;
  for (int j = 0; j < 16; ++j) { sc[j] = expf(sc[j] - mx); den += sc[j]; }
  const float inv = 1.0f / den;
  for (int e = 0; e < 8; ++e) {
    float o = 0.f;
    for (int j = 0; j < 16; ++j) o += sc[j] * (float)vb[rb + (long)j * 1024 + e];
    ob[rb + (long)qi * 1024 + e] = (_Float16)(o * inv);
  }
}

// ---------------- block(128)-wide sum ----------------
__device__ __forceinline__ float block128_sum(float v, float* red) {
  for (int o = 16; o > 0; o >>= 1) v += __shfl_down(v, o, 32);
  const int lane = threadIdx.x & 31, w = threadIdx.x >> 5;
  if (lane == 0) red[w] = v;
  __syncthreads();
  const float s = red[0] + red[1] + red[2] + red[3];
  __syncthreads();
  return s;
}

// ---------------- fuse: out1 = LN1(x + intra_expand + inter_gather) ----------------
__global__ __launch_bounds__(128) void k_fuse_ln1(
    const float* __restrict__ x, const float* __restrict__ xa2,
    const float* __restrict__ ilw, const float* __restrict__ ilb,
    const float* __restrict__ interb, const float* __restrict__ gg,
    const float* __restrict__ bb, float* __restrict__ out_f,
    _Float16* __restrict__ out_h) {
  const int row = blockIdx.x;  // (b,s,n)
  const int nn = row & 31, s = (row >> 5) & 127, b = row >> 12;
  const int d = threadIdx.x;
  float v = x[(long)row * 128 + d];
  // intra: sum_p xa2[p,b,n,d] * intra_lin_w[p,s] + intra_lin_b[s]
  float intr = ilb[s];
  for (int p = 0; p < 16; ++p)
    intr += xa2[((long)(p * 32 + b) * 32 + nn) * 128 + d] * ilw[p * 128 + s];
  // inter: gather through (b,N,PN,DP)->(b,SEQ,N,D) reshape
  const int f  = (s * 32 + nn) * 128 + d;
  const int no = f >> 14, rem = f & 16383, pn = rem >> 10, dp = rem & 1023;
  const float inter = interb[(((long)(b * 32 + no) * 16 + pn) << 10) + dp];
  v += intr + inter;
  __shared__ float red[4];
  const float mean = block128_sum(v, red) * (1.0f / 128.0f);
  const float dv   = v - mean;
  const float var  = block128_sum(dv * dv, red) * (1.0f / 128.0f);
  const float r = dv * rsqrtf(var + 1e-5f) * gg[d] + bb[d];
  out_f[(long)row * 128 + d] = r;
  out_h[(long)row * 128 + d] = (_Float16)r;
}

// ---------------- fuse: out = LN2(ffn_out + out1) ----------------
__global__ __launch_bounds__(128) void k_fuse_ln2(const float* __restrict__ ff,
                                                  const float* __restrict__ out1,
                                                  const float* __restrict__ gg,
                                                  const float* __restrict__ bb,
                                                  float* __restrict__ out) {
  const int row = blockIdx.x;
  const int d   = threadIdx.x;
  float v = ff[(long)row * 128 + d] + out1[(long)row * 128 + d];
  __shared__ float red[4];
  const float mean = block128_sum(v, red) * (1.0f / 128.0f);
  const float dv   = v - mean;
  const float var  = block128_sum(dv * dv, red) * (1.0f / 128.0f);
  out[(long)row * 128 + d] = dv * rsqrtf(var + 1e-5f) * gg[d] + bb[d];
}

// ==================================================================
extern "C" void kernel_launch(void* const* d_in, const int* in_sizes, int n_in,
                              void* d_out, int out_size, void* d_ws,
                              size_t ws_size, hipStream_t stream) {
  (void)in_sizes; (void)n_in; (void)out_size; (void)ws_size;
  const float* x      = (const float*)d_in[0];
  const float* ie     = (const float*)d_in[1];
  const float* egw    = (const float*)d_in[2];
  const float* egb    = (const float*)d_in[3];
  const float* memory = (const float*)d_in[4];
  const float* g1w = (const float*)d_in[5],  *g1b = (const float*)d_in[6];
  const float* g2w = (const float*)d_in[7],  *g2b = (const float*)d_in[8];
  const float* g3w = (const float*)d_in[9],  *g3b = (const float*)d_in[10];
  const float* Pd  = (const float*)d_in[11], *Qd  = (const float*)d_in[12];
  const float* Bd  = (const float*)d_in[13];
  const float* Ps  = (const float*)d_in[14], *Bs  = (const float*)d_in[15];
  const float* ilw = (const float*)d_in[16], *ilb = (const float*)d_in[17];
  const float* emb_w = (const float*)d_in[18], *emb_b = (const float*)d_in[19];
  const float* W_pos = (const float*)d_in[20];
  const float* wq = (const float*)d_in[21], *bq = (const float*)d_in[22];
  const float* wk = (const float*)d_in[23], *bk = (const float*)d_in[24];
  const float* wv = (const float*)d_in[25], *bv = (const float*)d_in[26];
  const float* wo = (const float*)d_in[27], *bo = (const float*)d_in[28];
  const float* ln1g = (const float*)d_in[29], *ln1b = (const float*)d_in[30];
  const float* fw1 = (const float*)d_in[31], *fb1 = (const float*)d_in[32];
  const float* fw2 = (const float*)d_in[33], *fb2 = (const float*)d_in[34];
  const float* ln2g = (const float*)d_in[35], *ln2b = (const float*)d_in[36];
  float* out = (float*)d_out;

  // ---- workspace bump allocator (deterministic layout every call) ----
  char* wp = (char*)d_ws;
  auto alloc = [&](size_t bytes) -> void* {
    void* r = (void*)wp;
    wp += (bytes + 255) & ~(size_t)255;
    return r;
  };
  _Float16* emb_w_h = (_Float16*)alloc((size_t)DP_ * DP_ * 2);
  _Float16* wq_h    = (_Float16*)alloc((size_t)DP_ * DP_ * 2);
  _Float16* wk_h    = (_Float16*)alloc((size_t)DP_ * DP_ * 2);
  _Float16* wv_h    = (_Float16*)alloc((size_t)DP_ * DP_ * 2);
  _Float16* wo_h    = (_Float16*)alloc((size_t)DP_ * DP_ * 2);
  _Float16* fw1_h   = (_Float16*)alloc((size_t)D_ * DFF_ * 2);
  _Float16* fw2_h   = (_Float16*)alloc((size_t)DFF_ * D_ * 2);
  _Float16* Ps_h    = (_Float16*)alloc((size_t)2 * D_ * D_ * 2);
  float*    mem_f   = (float*)alloc((size_t)N_ * 100 * 4);
  float*    bias_d  = (float*)alloc((size_t)2 * N_ * D_ * 4);
  _Float16* emb_h   = (_Float16*)alloc((size_t)PN_ * N_ * D_ * 2);
  _Float16* Wd_h    = (_Float16*)alloc((size_t)2 * N_ * D_ * D_ * 2);
  const long T_TOT  = (long)PN_ * B_ * 9 * N_ * D_;           // 18,874,368
  _Float16* t_h     = (_Float16*)alloc((size_t)T_TOT * 2);
  _Float16* kbuf_h  = (_Float16*)alloc((size_t)T_TOT * 2);
  _Float16* vbuf_h  = (_Float16*)alloc((size_t)T_TOT * 2);
  const long XA_TOT = (long)PN_ * B_ * N_ * D_;               // 2,097,152
  _Float16* xa_h    = (_Float16*)alloc((size_t)XA_TOT * 2);
  _Float16* xs_h    = (_Float16*)alloc((size_t)XA_TOT * 2);
  float*    xa2_f   = (float*)alloc((size_t)XA_TOT * 4);
  const long XR_TOT = (long)B_ * N_ * PN_ * DP_;              // 16,777,216
  _Float16* xr_h    = (_Float16*)alloc((size_t)XR_TOT * 2);
  _Float16* xe_h    = (_Float16*)alloc((size_t)XR_TOT * 2);
  _Float16* q_h     = (_Float16*)alloc((size_t)XR_TOT * 2);
  _Float16* k_h     = (_Float16*)alloc((size_t)XR_TOT * 2);
  _Float16* v_h     = (_Float16*)alloc((size_t)XR_TOT * 2);
  _Float16* oatt_h  = (_Float16*)alloc((size_t)XR_TOT * 2);
  float*    inter_f = (float*)alloc((size_t)XR_TOT * 4);
  const long ROWS   = (long)B_ * SEQ_ * N_;                   // 131,072
  float*    out1_f  = (float*)alloc((size_t)ROWS * D_ * 4);
  _Float16* out1_h  = (_Float16*)alloc((size_t)ROWS * D_ * 2);
  _Float16* hh_h    = (_Float16*)alloc((size_t)ROWS * DFF_ * 2);
  float*    ff_f    = (float*)alloc((size_t)ROWS * D_ * 4);

  // ---- 1) weight conversions f32 -> f16 ----
  k_cvt_f16<<<2048, 256, 0, stream>>>(emb_w, emb_w_h, (long)DP_ * DP_);
  k_cvt_f16<<<2048, 256, 0, stream>>>(wq, wq_h, (long)DP_ * DP_);
  k_cvt_f16<<<2048, 256, 0, stream>>>(wk, wk_h, (long)DP_ * DP_);
  k_cvt_f16<<<2048, 256, 0, stream>>>(wv, wv_h, (long)DP_ * DP_);
  k_cvt_f16<<<2048, 256, 0, stream>>>(wo, wo_h, (long)DP_ * DP_);
  k_cvt_f16<<<256, 256, 0, stream>>>(fw1, fw1_h, (long)D_ * DFF_);
  k_cvt_f16<<<256, 256, 0, stream>>>(fw2, fw2_h, (long)DFF_ * D_);
  k_cvt_f16<<<128, 256, 0, stream>>>(Ps, Ps_h, (long)2 * D_ * D_);

  // ---- 2) memory MLP, bias_d, Wd, emb ----
  k_prep_mem<<<1, 256, 0, stream>>>(memory, g1w, g1b, g2w, g2b, g3w, g3b, Bd,
                                    ie, egw, egb, mem_f, bias_d, emb_h);
  k_prep_wd<<<64, 256, 0, stream>>>(Pd, Qd, mem_f, Wd_h);

  // ---- 3) activation packing ----
  k_build_t<<<4096, 256, 0, stream>>>(x, emb_h, t_h, T_TOT);
  k_pack_xr<<<4096, 256, 0, stream>>>(x, xr_h, XR_TOT);

  // ---- 4) intra path: per-node K/V projections (batched over n=32) ----
  // A = t (rows p*b*tt = 4608, lda=4096, batch offset n*128), B = Wd[w][n]
  k_wmma_gemm<0, 1><<<dim3(1, 36, 32), 256, 0, stream>>>(
      t_h, 128, 4096, Wd_h + 0 * (long)N_ * 16384, 16384, 128,
      bias_d + 0 * (long)N_ * D_, 128, nullptr, 1, kbuf_h, 128, 4096, 4608, 128, 128);
  k_wmma_gemm<0, 1><<<dim3(1, 36, 32), 256, 0, stream>>>(
      t_h, 128, 4096, Wd_h + 1 * (long)N_ * 16384, 16384, 128,
      bias_d + 1 * (long)N_ * D_, 128, nullptr, 1, vbuf_h, 128, 4096, 4608, 128, 128);
  k_intra_attn<<<128, 256, 0, stream>>>(emb_h, kbuf_h, vbuf_h, xa_h);
  // Ps0 (+Bs0, relu) then Ps1 (+Bs1)
  k_wmma_gemm<1, 1><<<dim3(1, 128, 1), 256, 0, stream>>>(
      xa_h, 0, 128, Ps_h, 0, 128, Bs, 0, nullptr, 1, xs_h, 0, 128, 16384, 128, 128);
  k_wmma_gemm<0, 0><<<dim3(1, 128, 1), 256, 0, stream>>>(
      xs_h, 0, 128, Ps_h + (long)D_ * D_, 0, 128, Bs + D_, 0, nullptr, 1,
      xa2_f, 0, 128, 16384, 128, 128);

  // ---- 5) inter path ----
  // xe = xr @ emb_w + emb_b + W_pos[row%16]
  k_wmma_gemm<0, 1><<<dim3(8, 128, 1), 256, 0, stream>>>(
      xr_h, 0, 1024, emb_w_h, 0, 1024, emb_b, 0, W_pos, 16, xe_h, 0, 1024,
      16384, 1024, 1024);
  k_wmma_gemm<0, 1><<<dim3(8, 128, 1), 256, 0, stream>>>(
      xe_h, 0, 1024, wq_h, 0, 1024, bq, 0, nullptr, 1, q_h, 0, 1024, 16384, 1024, 1024);
  k_wmma_gemm<0, 1><<<dim3(8, 128, 1), 256, 0, stream>>>(
      xe_h, 0, 1024, wk_h, 0, 1024, bk, 0, nullptr, 1, k_h, 0, 1024, 16384, 1024, 1024);
  k_wmma_gemm<0, 1><<<dim3(8, 128, 1), 256, 0, stream>>>(
      xe_h, 0, 1024, wv_h, 0, 1024, bv, 0, nullptr, 1, v_h, 0, 1024, 16384, 1024, 1024);
  k_inter_attn<<<8192, 256, 0, stream>>>(q_h, k_h, v_h, oatt_h);
  k_wmma_gemm<0, 0><<<dim3(8, 128, 1), 256, 0, stream>>>(
      oatt_h, 0, 1024, wo_h, 0, 1024, bo, 0, nullptr, 1, inter_f, 0, 1024,
      16384, 1024, 1024);

  // ---- 6) residual + LN1 ----
  k_fuse_ln1<<<(int)ROWS, 128, 0, stream>>>(x, xa2_f, ilw, ilb, inter_f, ln1g,
                                            ln1b, out1_f, out1_h);

  // ---- 7) FFN (gelu) + LN2 ----
  k_wmma_gemm<2, 1><<<dim3(4, 1024, 1), 256, 0, stream>>>(
      out1_h, 0, 128, fw1_h, 0, 512, fb1, 0, nullptr, 1, hh_h, 0, 512,
      (int)ROWS, 512, 128);
  k_wmma_gemm<0, 0><<<dim3(1, 1024, 1), 256, 0, stream>>>(
      hh_h, 0, 512, fw2_h, 0, 128, fb2, 0, nullptr, 1, ff_f, 0, 128,
      (int)ROWS, 128, 512);
  k_fuse_ln2<<<(int)ROWS, 128, 0, stream>>>(ff_f, out1_f, ln2g, ln2b, out);
}
